// DiffAttention_12249246728639
// MI455X (gfx1250) — matrix operand order repaired and checked by reading, not compiled, not run on hardware
//
#include <hip/hip_runtime.h>
#include <stdint.h>

// ---------------- problem constants ----------------
#define S_LEN 2048
#define HIDN  2048
#define NH    16
#define NKV   8
#define HDIM  128
#define VDIM  256               // 2*HDIM per kv head
#define LAMBDA_INIT 0.2f
#define EPSV 1e-5f
#define ATT_SCALE 0.088388347648318447f   // 128^-0.5

// ---------------- types ----------------
typedef unsigned short u16;
typedef __attribute__((ext_vector_type(8)))  u16    v8u;
typedef __attribute__((ext_vector_type(16))) u16    v16u;
typedef __attribute__((ext_vector_type(16))) __bf16 v16bf;
typedef __attribute__((ext_vector_type(8)))  float  v8f;

// fp32 -> bf16 bits, round-to-nearest-even
__device__ __forceinline__ u16 f2bf_bits(float f) {
    union { float f; uint32_t u; } a; a.f = f;
    uint32_t u = a.u;
    uint32_t r = (u + 0x7fffu + ((u >> 16) & 1u)) >> 16;
    return (u16)r;
}

// A fragment: 16x32 bf16 (MxK). p -> element (row0, k0); ld = row stride (elems).
// lanes 0-15: M=0..15, K chunks [0..7] and [16..23]; lanes 16-31: K chunks [8..15],[24..31]
__device__ __forceinline__ v16bf frag_a(const u16* p, size_t ld, int lane) {
    int m  = lane & 15;
    int hi = (lane >> 4) << 3;
    const u16* q = p + (size_t)m * ld;
    v8u c0 = *(const v8u*)(q + hi);
    v8u c1 = *(const v8u*)(q + 16 + hi);
    v16u r;
#pragma unroll
    for (int i = 0; i < 8; ++i) { r[i] = c0[i]; r[8 + i] = c1[i]; }
    return __builtin_bit_cast(v16bf, r);
}

// B fragment: 32x16 bf16 (KxN), column n contiguous along K in memory.
// p -> element (k0, col0); ld = stride between columns (elems).
// lanes 0-15 hold K [0..7],[16..23] of col lane&15; lanes 16-31 hold K [8..15],[24..31].
__device__ __forceinline__ v16bf frag_b(const u16* p, size_t ld, int lane) {
    const u16* q = p + (size_t)(lane & 15) * ld + ((lane >> 4) << 3);
    v8u c0 = *(const v8u*)(q);
    v8u c1 = *(const v8u*)(q + 16);
    v16u r;
#pragma unroll
    for (int i = 0; i < 8; ++i) { r[i] = c0[i]; r[8 + i] = c1[i]; }
    return __builtin_bit_cast(v16bf, r);
}

__device__ __forceinline__ v8f wmma_bf16(v16bf a, v16bf b, v8f c) {
    return __builtin_amdgcn_wmma_f32_16x16x32_bf16(false, a, false, b, (short)0, c, false, false);
}

// ---------------- elementwise fp32 -> bf16 cast ----------------
__global__ void k_cast_bf16(const float* __restrict__ in, u16* __restrict__ out, size_t n) {
    size_t i = (size_t)blockIdx.x * blockDim.x + threadIdx.x;
    if (i < n) out[i] = f2bf_bits(in[i]);
}

// ---------------- V transpose + cast: v (S, 2048) -> vT (2048, S) ----------------
__global__ void k_vtrans(const float* __restrict__ v, u16* __restrict__ vT) {
    size_t i = (size_t)blockIdx.x * blockDim.x + threadIdx.x;
    if (i >= (size_t)S_LEN * (NKV * VDIM)) return;
    int f = (int)(i >> 11);          // feature row of vT
    int s = (int)(i & 2047);         // sequence col
    vT[i] = f2bf_bits(v[(size_t)s * (NKV * VDIM) + f]);
}

// ---------------- RoPE + cast to bf16 (q/k/qn/kn) ----------------
__global__ void k_rope_cast(const float* __restrict__ in, const float* __restrict__ cosp,
                            const float* __restrict__ sinp, u16* __restrict__ out, int heads) {
    size_t idx = (size_t)blockIdx.x * blockDim.x + threadIdx.x;
    size_t n = (size_t)S_LEN * heads * HDIM;
    if (idx >= n) return;
    int d = (int)(idx & (HDIM - 1));
    int h = (int)((idx >> 7) % heads);
    int s = (int)(idx / ((size_t)heads * HDIM));
    const float* row = in + ((size_t)s * heads + h) * HDIM;
    float x  = row[d];
    float xr = (d < HDIM / 2) ? -row[d + HDIM / 2] : row[d - HDIM / 2];
    float r  = x * cosp[(size_t)s * HDIM + d] + xr * sinp[(size_t)s * HDIM + d];
    out[idx] = f2bf_bits(r);
}

// ---------------- per-head lambda ----------------
__global__ void k_lambda(const float* __restrict__ lq1, const float* __restrict__ lk1,
                         const float* __restrict__ lq2, const float* __restrict__ lk2,
                         float* __restrict__ lam) {
    int h = threadIdx.x;
    if (h >= NH) return;
    float s1 = 0.f, s2 = 0.f;
    for (int d = 0; d < HDIM; ++d) {
        s1 += lq1[h * HDIM + d] * lk1[h * HDIM + d];
        s2 += lq2[h * HDIM + d] * lk2[h * HDIM + d];
    }
    lam[h] = __expf(s1) - __expf(s2) + LAMBDA_INIT;
}

// ---------------- bf16 WMMA GEMM: C(M,N) = A(M,K) @ B(N,K)^T + bias ----------------
// block = 256 threads (8 waves), block tile 32(M) x 128(N), wave tile 16x32
// (two 16x16 accumulators share one A fragment -> 2 WMMA per A load).
__global__ __launch_bounds__(256)
void k_gemm_bf16(const u16* __restrict__ A, const u16* __restrict__ Bw,
                 const float* __restrict__ bias, float* __restrict__ C,
                 int M, int N, int K) {
    int wave = threadIdx.x >> 5;
    int lane = threadIdx.x & 31;
    int m0 = blockIdx.y * 32  + (wave >> 2) * 16;
    int n0 = blockIdx.x * 128 + (wave & 3) * 32;
    const u16* arow  = A  + (size_t)m0 * K;
    const u16* b0row = Bw + (size_t)n0 * K;
    const u16* b1row = Bw + (size_t)(n0 + 16) * K;
    v8f acc0 = {}, acc1 = {};
    for (int k0 = 0; k0 < K; k0 += 32) {
        v16bf a  = frag_a(arow + k0, K, lane);
        v16bf b0 = frag_b(b0row + k0, K, lane);
        v16bf b1 = frag_b(b1row + k0, K, lane);
        acc0 = wmma_bf16(a, b0, acc0);
        acc1 = wmma_bf16(a, b1, acc1);
    }
    int col   = n0 + (lane & 15);
    int rbase = (lane >> 4) * 8;
    float bv0 = bias ? bias[col]      : 0.f;
    float bv1 = bias ? bias[col + 16] : 0.f;
#pragma unroll
    for (int i = 0; i < 8; ++i) {
        C[(size_t)(m0 + rbase + i) * N + col]      = acc0[i] + bv0;
        C[(size_t)(m0 + rbase + i) * N + col + 16] = acc1[i] + bv1;
    }
}

// ---------------- fused differential attention ----------------
// grid (NH, S/16); block 256 (8 waves).
// LDS: two 16x2048 fp32 score strips (256 KB dynamic) -> fits CDNA5's 320 KB WGP LDS.
__global__ __launch_bounds__(256)
void k_attn(const u16* __restrict__ qbf, const u16* __restrict__ kbf,
            const u16* __restrict__ qnbf, const u16* __restrict__ knbf,
            const u16* __restrict__ vT, const float* __restrict__ lam,
            float* __restrict__ attn_out, float* __restrict__ out_h) {
    extern __shared__ char smemraw[];
    float* s1 = (float*)smemraw;          // 16*2048 f32: branch-1 scores/probs
    float* s2 = s1 + 16 * S_LEN;          // 16*2048 f32: branch-2 scores/probs
    __shared__ float red[256];
    __shared__ float rstat[4][16];        // max1,sum1,max2,sum2 per row

    const int h    = blockIdx.x;
    const int q0   = blockIdx.y * 16;
    const int kv   = h >> 1;
    const int tid  = threadIdx.x;
    const int wave = tid >> 5;
    const int lane = tid & 31;

    // ---- phase 1: S1 = Q K^T * scale, S2 = Qn Kn^T * scale (WMMA bf16) ----
    // Q fragments (A operand) depend only on k0 -> hoist all 4 out of the column loop.
    for (int which = 0; which < 2; ++which) {
        const u16* Q  = which ? qnbf : qbf;
        const u16* Kp = which ? knbf : kbf;
        float* S      = which ? s2 : s1;
        const u16* qrow = Q + (size_t)q0 * (NH * HDIM) + h * HDIM;
        v16bf aq[4];
#pragma unroll
        for (int kk = 0; kk < 4; ++kk)
            aq[kk] = frag_a(qrow + kk * 32, NH * HDIM, lane);
        const int n0w = wave * 256;
        for (int t = 0; t < 16; ++t) {
            int n0 = n0w + t * 16;
            const u16* kbase = Kp + (size_t)n0 * (NKV * HDIM) + kv * HDIM;
            v8f acc = {};
#pragma unroll
            for (int kk = 0; kk < 4; ++kk) {
                v16bf b = frag_b(kbase + kk * 32, NKV * HDIM, lane);
                acc = wmma_bf16(aq[kk], b, acc);
            }
            int col   = n0 + (lane & 15);
            int rbase = (lane >> 4) * 8;
#pragma unroll
            for (int i = 0; i < 8; ++i)
                S[(size_t)(rbase + i) * S_LEN + col] = acc[i] * ATT_SCALE;
        }
    }
    __syncthreads();

    // ---- phase 2: exact row softmax for both branches (16 threads/row x 128 elems) ----
    const int r = tid & 15;
    const int c = tid >> 4;
    for (int which = 0; which < 2; ++which) {
        float* S = which ? s2 : s1;
        float* p = S + (size_t)r * S_LEN + c * 128;
        float lm = -3.0e38f;
        for (int j = 0; j < 128; ++j) lm = fmaxf(lm, p[j]);
        red[tid] = lm;
        __syncthreads();
        if (c == 0) {
            float m = red[r];
            for (int cc = 1; cc < 16; ++cc) m = fmaxf(m, red[cc * 16 + r]);
            rstat[which * 2][r] = m;
        }
        __syncthreads();
        float rm = rstat[which * 2][r];
        float ls = 0.f;
        for (int j = 0; j < 128; ++j) { float e = __expf(p[j] - rm); p[j] = e; ls += e; }
        red[tid] = ls;
        __syncthreads();
        if (c == 0) {
            float s = red[r];
            for (int cc = 1; cc < 16; ++cc) s += red[cc * 16 + r];
            rstat[which * 2 + 1][r] = s;
        }
        __syncthreads();
    }

    // ---- phase 3: attn = P1 - lam*P2; write global attn output; keep fp32 in s1 ----
    {
        const float lamh = lam[h];
        const float inv1 = 1.f / rstat[1][r];
        const float inv2 = 1.f / rstat[3][r];
        float* p1 = s1 + (size_t)r * S_LEN + c * 128;
        float* p2 = s2 + (size_t)r * S_LEN + c * 128;
        float* g  = attn_out + ((size_t)h * S_LEN + (q0 + r)) * S_LEN + c * 128;
        for (int j = 0; j < 128; ++j) {
            float a = p1[j] * inv1 - lamh * (p2[j] * inv2);
            p1[j] = a;
            g[j]  = a;
        }
    }
    __syncthreads();
    // re-quantize attn to bf16 into s2's storage (disjoint buffer -> no aliasing race)
    u16* pb = (u16*)s2;
    {
        u16*   dst = pb + (size_t)r * S_LEN + c * 128;
        float* src = s1 + (size_t)r * S_LEN + c * 128;
        for (int j = 0; j < 128; ++j) dst[j] = f2bf_bits(src[j]);
    }
    __syncthreads();

    // ---- phase 4: out(16x256) = attn(16x2048) @ V_h(2048x256) via WMMA ----
    const int f0 = wave * 32;                 // each wave: 2 column tiles sharing A
    const u16* vbase = vT + (size_t)(kv * VDIM + f0) * S_LEN;
    v8f acc0 = {}, acc1 = {};
    for (int k0 = 0; k0 < S_LEN; k0 += 32) {
        v16bf a  = frag_a(pb + k0, S_LEN, lane);
        v16bf b0 = frag_b(vbase + k0, S_LEN, lane);
        v16bf b1 = frag_b(vbase + (size_t)16 * S_LEN + k0, S_LEN, lane);
        acc0 = wmma_bf16(a, b0, acc0);
        acc1 = wmma_bf16(a, b1, acc1);
    }
    const int rbase = (lane >> 4) * 8;
    const int fc    = lane & 15;
#pragma unroll
    for (int i = 0; i < 8; ++i) {
        size_t row = (size_t)h * S_LEN + q0 + rbase + i;
        out_h[row * VDIM + f0 + fc]      = acc0[i];
        out_h[row * VDIM + f0 + 16 + fc] = acc1[i];
    }
}

// ---------------- group statistics (32 groups of 262144 contiguous elems) ----------------
__global__ void k_gstats(const float* __restrict__ out_h, float* __restrict__ stats) {
    const int g = blockIdx.x;
    const int tid = threadIdx.x;
    const float* base = out_h + (size_t)g * 262144;
    float s = 0.f, ss = 0.f;
    for (size_t i = tid; i < 262144; i += 256) { float v = base[i]; s += v; ss += v * v; }
    __shared__ float rs[256], rss[256];
    rs[tid] = s; rss[tid] = ss;
    __syncthreads();
    for (int o = 128; o > 0; o >>= 1) {
        if (tid < o) { rs[tid] += rs[tid + o]; rss[tid] += rss[tid + o]; }
        __syncthreads();
    }
    if (tid == 0) {
        float mu  = rs[0]  * (1.f / 262144.f);
        float var = rss[0] * (1.f / 262144.f) - mu * mu;
        stats[g * 2]     = mu;
        stats[g * 2 + 1] = var;
    }
}

// ---------------- normalize + gamma/beta + (1-lambda_init) + transpose to (S, 4096) bf16 ----------------
__global__ void k_norm_cast(const float* __restrict__ out_h, const float* __restrict__ stats,
                            const float* __restrict__ gamma, const float* __restrict__ beta,
                            u16* __restrict__ ybf) {
    size_t idx = (size_t)blockIdx.x * blockDim.x + threadIdx.x;
    if (idx >= (size_t)NH * S_LEN * VDIM) return;
    int h = (int)(idx >> 19);
    int q = (int)((idx >> 8) & 2047);
    int d = (int)(idx & 255);
    int half   = q >> 10;
    int g      = (h << 1) | half;
    int inflat = (int)(idx & 262143);
    int gd     = inflat >> 11;                 // gamma/beta index (reference's reshape semantics)
    float mu  = stats[g * 2];
    float var = stats[g * 2 + 1];
    float v = out_h[idx];
    float y = (v - mu) * rsqrtf(var + EPSV) * gamma[gd] + beta[gd];
    y *= (1.0f - LAMBDA_INIT);
    ybf[(size_t)q * (2 * NH * HDIM) + h * VDIM + d] = f2bf_bits(y);
}

// ---------------- host-side launch ----------------
extern "C" void kernel_launch(void* const* d_in, const int* in_sizes, int n_in,
                              void* d_out, int out_size, void* d_ws, size_t ws_size,
                              hipStream_t stream) {
    const float* X    = (const float*)d_in[0];
    const float* cosp = (const float*)d_in[1];
    const float* sinp = (const float*)d_in[2];
    const float* Wq   = (const float*)d_in[3];
    const float* bq   = (const float*)d_in[4];
    const float* Wk   = (const float*)d_in[5];
    const float* bk   = (const float*)d_in[6];
    const float* Wqn  = (const float*)d_in[7];
    const float* bqn  = (const float*)d_in[8];
    const float* Wkn  = (const float*)d_in[9];
    const float* bkn  = (const float*)d_in[10];
    const float* Wv   = (const float*)d_in[11];
    const float* bv   = (const float*)d_in[12];
    const float* Wo   = (const float*)d_in[13];
    const float* bo   = (const float*)d_in[14];
    const float* lq1  = (const float*)d_in[15];
    const float* lk1  = (const float*)d_in[16];
    const float* lq2  = (const float*)d_in[17];
    const float* lk2  = (const float*)d_in[18];
    const float* gam  = (const float*)d_in[19];
    const float* bet  = (const float*)d_in[20];

    float* out      = (float*)d_out;                              // (S, 2048)
    float* attn_out = out + (size_t)S_LEN * HIDN;                 // (H, S, S)

    // workspace cursor
    char* ws = (char*)d_ws;
    size_t off = 0;
    auto alloc = [&](size_t bytes) -> char* {
        char* p = ws + off;
        off = (off + bytes + 255) & ~(size_t)255;
        return p;
    };
    const size_t NE_X  = (size_t)S_LEN * HIDN;       // 4.19M
    const size_t NE_Wq = (size_t)NH * HDIM * HIDN;   // 4.19M
    const size_t NE_Wk = (size_t)NKV * HDIM * HIDN;  // 2.10M
    const size_t NE_Wv = (size_t)NKV * VDIM * HIDN;  // 4.19M
    const size_t NE_Wo = (size_t)HIDN * (2 * HIDN);  // 8.39M

    u16* Xbf   = (u16*)alloc(NE_X  * 2);
    u16* Wqbf  = (u16*)alloc(NE_Wq * 2);
    u16* Wkbf  = (u16*)alloc(NE_Wk * 2);
    u16* Wqnbf = (u16*)alloc(NE_Wq * 2);
    u16* Wknbf = (u16*)alloc(NE_Wk * 2);
    u16* Wvbf  = (u16*)alloc(NE_Wv * 2);
    u16* Wobf  = (u16*)alloc(NE_Wo * 2);
    float* qf  = (float*)alloc((size_t)S_LEN * NH * HDIM * 4);
    float* kf  = (float*)alloc((size_t)S_LEN * NKV * HDIM * 4);
    float* qnf = (float*)alloc((size_t)S_LEN * NH * HDIM * 4);
    float* knf = (float*)alloc((size_t)S_LEN * NKV * HDIM * 4);
    float* vf  = (float*)alloc((size_t)S_LEN * NKV * VDIM * 4);
    u16* qbf   = (u16*)alloc((size_t)S_LEN * NH * HDIM * 2);
    u16* kbf   = (u16*)alloc((size_t)S_LEN * NKV * HDIM * 2);
    u16* qnbf  = (u16*)alloc((size_t)S_LEN * NH * HDIM * 2);
    u16* knbf  = (u16*)alloc((size_t)S_LEN * NKV * HDIM * 2);
    u16* vTbf  = (u16*)alloc((size_t)S_LEN * NKV * VDIM * 2);
    float* outh = (float*)alloc((size_t)NH * S_LEN * VDIM * 4);
    u16* ybf   = (u16*)alloc((size_t)S_LEN * 2 * NH * HDIM * 2);
    float* lam = (float*)alloc(64);
    float* stats = (float*)alloc(32 * 2 * 4);

    const int TB = 256;
    auto blocks = [&](size_t n) { return (unsigned)((n + TB - 1) / TB); };

    // 1) fp32 -> bf16 casts (activations + weights; weights all fit in the 192MB L2)
    k_cast_bf16<<<blocks(NE_X),  TB, 0, stream>>>(X,   Xbf,   NE_X);
    k_cast_bf16<<<blocks(NE_Wq), TB, 0, stream>>>(Wq,  Wqbf,  NE_Wq);
    k_cast_bf16<<<blocks(NE_Wk), TB, 0, stream>>>(Wk,  Wkbf,  NE_Wk);
    k_cast_bf16<<<blocks(NE_Wq), TB, 0, stream>>>(Wqn, Wqnbf, NE_Wq);
    k_cast_bf16<<<blocks(NE_Wk), TB, 0, stream>>>(Wkn, Wknbf, NE_Wk);
    k_cast_bf16<<<blocks(NE_Wv), TB, 0, stream>>>(Wv,  Wvbf,  NE_Wv);
    k_cast_bf16<<<blocks(NE_Wo), TB, 0, stream>>>(Wo,  Wobf,  NE_Wo);

    // 2) projection GEMMs (bf16 WMMA, fp32 accum); block tile 32x128
    k_gemm_bf16<<<dim3((NH * HDIM) / 128,  S_LEN / 32), TB, 0, stream>>>(Xbf, Wqbf,  bq,  qf,  S_LEN, NH * HDIM,  HIDN);
    k_gemm_bf16<<<dim3((NKV * HDIM) / 128, S_LEN / 32), TB, 0, stream>>>(Xbf, Wkbf,  bk,  kf,  S_LEN, NKV * HDIM, HIDN);
    k_gemm_bf16<<<dim3((NH * HDIM) / 128,  S_LEN / 32), TB, 0, stream>>>(Xbf, Wqnbf, bqn, qnf, S_LEN, NH * HDIM,  HIDN);
    k_gemm_bf16<<<dim3((NKV * HDIM) / 128, S_LEN / 32), TB, 0, stream>>>(Xbf, Wknbf, bkn, knf, S_LEN, NKV * HDIM, HIDN);
    k_gemm_bf16<<<dim3((NKV * VDIM) / 128, S_LEN / 32), TB, 0, stream>>>(Xbf, Wvbf,  bv,  vf,  S_LEN, NKV * VDIM, HIDN);

    // 3) RoPE + bf16 cast; V transpose+cast (so B-fragment columns are contiguous)
    k_rope_cast<<<blocks((size_t)S_LEN * NH * HDIM),  TB, 0, stream>>>(qf,  cosp, sinp, qbf,  NH);
    k_rope_cast<<<blocks((size_t)S_LEN * NKV * HDIM), TB, 0, stream>>>(kf,  cosp, sinp, kbf,  NKV);
    k_rope_cast<<<blocks((size_t)S_LEN * NH * HDIM),  TB, 0, stream>>>(qnf, cosp, sinp, qnbf, NH);
    k_rope_cast<<<blocks((size_t)S_LEN * NKV * HDIM), TB, 0, stream>>>(knf, cosp, sinp, knbf, NKV);
    k_vtrans<<<blocks((size_t)S_LEN * NKV * VDIM), TB, 0, stream>>>(vf, vTbf);

    // 4) per-head lambda
    k_lambda<<<1, 32, 0, stream>>>(lq1, lk1, lq2, lk2, lam);

    // 5) fused differential attention: scores + dual softmax + combine + PV
    //    dynamic LDS = 2 * 16 * 2048 * 4 = 256 KB (within CDNA5's 320 KB/WGP)
    k_attn<<<dim3(NH, S_LEN / 16), TB, 2 * 16 * S_LEN * sizeof(float), stream>>>(
        qbf, kbf, qnbf, knbf, vTbf, lam, attn_out, outh);

    // 6) group statistics + normalization (reference's (B*H*2, HD, S) reshape semantics)
    k_gstats<<<32, TB, 0, stream>>>(outh, stats);
    k_norm_cast<<<blocks((size_t)NH * S_LEN * VDIM), TB, 0, stream>>>(outh, stats, gam, bet, ybf);

    // 7) output projection: (S,4096) @ Wo^T (2048,4096) + bo -> d_out
    k_gemm_bf16<<<dim3(HIDN / 128, S_LEN / 32), TB, 0, stream>>>(ybf, Wobf, bo, out, S_LEN, HIDN, 2 * HIDN);
}
